// Attention_48859547959768
// MI455X (gfx1250) — matrix operand order repaired and checked by reading, not compiled, not run on hardware
//
#include <hip/hip_runtime.h>
#include <stdint.h>

// ---------------------------------------------------------------------------
// MI455X / gfx1250 multi-head attention forward, bf16 WMMA path with
// async-to-LDS double-buffered K/V staging in the attention kernel.
// ---------------------------------------------------------------------------

typedef __attribute__((ext_vector_type(16))) __bf16 bf16x16;
typedef __attribute__((ext_vector_type(8)))  float  f32x8;

static constexpr int B_   = 2;
static constexpr int S_   = 2048;
static constexpr int DIM_ = 1024;
static constexpr int H_   = 16;
static constexpr int D_   = 64;
static constexpr int M_   = B_ * S_;          // 4096 rows in all GEMMs

// ---- bf16 helpers (stored as ushort in workspace) -------------------------
__device__ __forceinline__ unsigned short f2bf(float f) {
    unsigned u = __builtin_bit_cast(unsigned, f);
    u += 0x7FFFu + ((u >> 16) & 1u);          // round-to-nearest-even
    return (unsigned short)(u >> 16);
}

__device__ __forceinline__ bf16x16 frag_from(uint4 lo, uint4 hi) {
    union { bf16x16 v; uint4 q[2]; } u;
    u.q[0] = lo; u.q[1] = hi;
    return u.v;
}

// A fragment (16x32 bf16, MxK). ISA layout: lane 0-15 = rows, two 8-elem K
// chunks: [k0+kb .. +7] and [k0+16+kb .. +7], kb = (lane>>4)*8.
__device__ __forceinline__ bf16x16 load_a_frag(const unsigned short* base,
                                               int ld, int row0, int k0) {
    const int lane = threadIdx.x & 31;
    const int r    = lane & 15;
    const int kb   = (lane >> 4) * 8;
    const unsigned short* p = base + (size_t)(row0 + r) * ld + k0;
    uint4 lo = *reinterpret_cast<const uint4*>(p + kb);
    uint4 hi = *reinterpret_cast<const uint4*>(p + 16 + kb);
    return frag_from(lo, hi);
}

// B fragment (32x16 bf16, KxN) loaded from B^T stored row-major [N][K]:
// lane = column N, two 8-elem K chunks: [k0+kb .. +7] and [k0+16+kb .. +7], kb = (lane>>4)*8.
__device__ __forceinline__ bf16x16 load_b_frag(const unsigned short* baseT,
                                               int ld, int n0, int k0) {
    const int lane = threadIdx.x & 31;
    const int n    = lane & 15;
    const int ko   = (lane >> 4) * 8;
    const unsigned short* p = baseT + (size_t)(n0 + n) * ld + k0 + ko;
    uint4 lo = *reinterpret_cast<const uint4*>(p);
    uint4 hi = *reinterpret_cast<const uint4*>(p + 16);
    return frag_from(lo, hi);
}

__device__ __forceinline__ f32x8 wmma_bf16(bf16x16 a, bf16x16 b, f32x8 c) {
    return __builtin_amdgcn_wmma_f32_16x16x32_bf16(
        /*neg_a=*/false, a, /*neg_b=*/false, b,
        /*c_mod=*/(short)0, c, /*reuse_a=*/false, /*reuse_b=*/false);
}

// Async 16-byte global -> LDS copy (per-lane addresses), tracked by ASYNCcnt.
__device__ __forceinline__ void async_copy_b128(unsigned ldsOff, const void* g) {
    asm volatile("global_load_async_to_lds_b128 %0, %1, off"
                 :: "v"(ldsOff), "v"(g) : "memory");
}

// ---------------------------------------------------------------------------
// Kernel 1: fp32 -> bf16 bulk convert (vector-4)
// ---------------------------------------------------------------------------
__global__ __launch_bounds__(256) void cvt_f32_bf16(const float* __restrict__ x,
                                                    unsigned short* __restrict__ y,
                                                    int n) {
    int i = (blockIdx.x * blockDim.x + threadIdx.x) * 4;
    if (i + 3 < n) {
        float4 v = *reinterpret_cast<const float4*>(x + i);
        uint2 o;
        o.x = (unsigned)f2bf(v.x) | ((unsigned)f2bf(v.y) << 16);
        o.y = (unsigned)f2bf(v.z) | ((unsigned)f2bf(v.w) << 16);
        *reinterpret_cast<uint2*>(y + i) = o;
    }
}

// ---------------------------------------------------------------------------
// Kernel 2: weight transpose + convert: W[K][N] fp32 -> Wt[N][K] bf16
// ---------------------------------------------------------------------------
__global__ __launch_bounds__(256) void wt_cvt_transpose(const float* __restrict__ W,
                                                        unsigned short* __restrict__ Wt) {
    constexpr int K = DIM_, N = DIM_;
    __shared__ float tile[16][17];
    const int bx = blockIdx.x & 63;   // n-tile
    const int by = blockIdx.x >> 6;   // k-tile
    const int tx = threadIdx.x & 15;
    const int ty = threadIdx.x >> 4;
    tile[ty][tx] = W[(size_t)(by * 16 + ty) * N + bx * 16 + tx];
    __syncthreads();
    Wt[(size_t)(bx * 16 + ty) * K + by * 16 + tx] = f2bf(tile[tx][ty]);
}

// ---------------------------------------------------------------------------
// Kernel 3: WMMA GEMM  out = A[M,K] * Bt[N,K]^T + bias
//   mode 0: bf16 head-major out  [B,H,S,D]   (Q / K)
//   mode 1: bf16 transposed out  [B,H,D,S]   (V -> Vt)
//   mode 2: fp32 flat out        [M,N]       (final projection)
// ---------------------------------------------------------------------------
__global__ __launch_bounds__(128) void gemm_bf16_wmma(const unsigned short* __restrict__ A,
                                                      const unsigned short* __restrict__ Bt,
                                                      const float* __restrict__ bias,
                                                      void* __restrict__ outp,
                                                      int mode) {
    constexpr int K = DIM_, N = DIM_;
    const int wave = blockIdx.x * (blockDim.x >> 5) + (threadIdx.x >> 5);
    const int ntiles = N / 64;                 // 16
    const int mt = wave / ntiles;
    const int nt = wave % ntiles;
    const int m0 = mt * 16;
    const int n0 = nt * 64;

    f32x8 acc[4] = {{0,0,0,0,0,0,0,0}, {0,0,0,0,0,0,0,0},
                    {0,0,0,0,0,0,0,0}, {0,0,0,0,0,0,0,0}};

    for (int k0 = 0; k0 < K; k0 += 32) {
        if (k0 + 32 < K)
            __builtin_prefetch(A + (size_t)(m0 + (threadIdx.x & 15)) * K + k0 + 32, 0, 3);
        bf16x16 a = load_a_frag(A, K, m0, k0);
#pragma unroll
        for (int j = 0; j < 4; ++j) {
            bf16x16 b = load_b_frag(Bt, K, n0 + j * 16, k0);
            acc[j] = wmma_bf16(a, b, acc[j]);
        }
    }

    const int lane = threadIdx.x & 31;
    const int nloc = lane & 15;
    const int mloc = (lane >> 4) * 8;
#pragma unroll
    for (int j = 0; j < 4; ++j) {
        const int n = n0 + j * 16 + nloc;
        const float bv = bias[n];
#pragma unroll
        for (int r = 0; r < 8; ++r) {
            const int m = m0 + mloc + r;
            const float v = acc[j][r] + bv;
            if (mode == 2) {
                reinterpret_cast<float*>(outp)[(size_t)m * N + n] = v;
            } else {
                const int b = m / S_, s = m % S_;
                const int h = n / D_, d = n % D_;
                unsigned short* o = reinterpret_cast<unsigned short*>(outp);
                if (mode == 0)
                    o[(size_t)((b * H_ + h) * S_ + s) * D_ + d] = f2bf(v);
                else
                    o[(size_t)((b * H_ + h) * D_ + d) * S_ + s] = f2bf(v);
            }
        }
    }
}

// ---------------------------------------------------------------------------
// Kernel 4: flash attention.
// One block = one (b,h) head x 4 query tiles (one 16-row tile per wave).
// Per 32-key step: K tile [32k x 64d] and V tile [64d x 32k] are staged into
// LDS cooperatively with global_load_async_to_lds_b128, double-buffered so
// the next tiles stream in (ASYNCcnt) while the current ones feed 8 WMMAs.
// ---------------------------------------------------------------------------
__global__ __launch_bounds__(128) void flash_attn(const unsigned short* __restrict__ Qh,
                                                  const unsigned short* __restrict__ Kh,
                                                  const unsigned short* __restrict__ Vt,
                                                  const float* __restrict__ mask,
                                                  unsigned short* __restrict__ Ctx) {
    __shared__ unsigned short kT[2][32 * 64];   // 2 x 4 KB  (rows = key, ld 64)
    __shared__ unsigned short vT[2][64 * 32];   // 2 x 4 KB  (rows = d,   ld 32)
    __shared__ unsigned short pT[4][16 * 32];   // 1 KB per wave (P transpose)

    const int widx = threadIdx.x >> 5;
    const int lane = threadIdx.x & 31;
    const int qgroups = S_ / 64;                // 32 blocks per head
    const int bh = blockIdx.x / qgroups;        // 0 .. B*H-1
    const int qg = blockIdx.x % qgroups;
    const int b  = bh / H_;
    const int h  = bh % H_;
    const int q0 = qg * 64 + widx * 16;

    const unsigned short* Qb = Qh + (size_t)bh * S_ * D_;
    const unsigned short* Kb = Kh + (size_t)bh * S_ * D_;
    const unsigned short* Vb = Vt + (size_t)bh * D_ * S_;
    const float* mrow = mask + (size_t)b * S_;

    const bf16x16 aQ0 = load_a_frag(Qb, D_, q0, 0);
    const bf16x16 aQ1 = load_a_frag(Qb, D_, q0, 32);

    f32x8 ctx[4] = {{0,0,0,0,0,0,0,0}, {0,0,0,0,0,0,0,0},
                    {0,0,0,0,0,0,0,0}, {0,0,0,0,0,0,0,0}};
    float rm[8], rs[8];
#pragma unroll
    for (int r = 0; r < 8; ++r) { rm[r] = -3.0e38f; rs[r] = 0.0f; }

    const int nloc  = lane & 15;
    const int mbase = (lane >> 4) * 8;
    unsigned short* myP = pT[widx];

    // Cooperative async staging of one 32-key K tile + V tile into buffer buf.
    // 256 16B chunks per tile, 128 threads -> 2 chunks each per tile.
    auto issue_tiles = [&](int buf, int kb) {
        const unsigned kOff = (unsigned)(uintptr_t)(&kT[buf][0]);
        const unsigned vOff = (unsigned)(uintptr_t)(&vT[buf][0]);
        const char* kSrc = (const char*)(Kb + (size_t)kb * D_);   // contiguous 4KB
        const char* vBase = (const char*)Vb;
#pragma unroll
        for (int t = 0; t < 2; ++t) {
            const int c = threadIdx.x * 2 + t;                    // 0..255
            async_copy_b128(kOff + c * 16, kSrc + c * 16);
            // V tile: d = c>>2 (row, ld S_), key chunk = c&3 (8 keys each)
            const size_t vByte = ((size_t)(c >> 2) * S_ + kb) * 2 + (size_t)(c & 3) * 16;
            async_copy_b128(vOff + c * 16, vBase + vByte);
        }
    };

    issue_tiles(0, 0);                          // prologue

    constexpr int NBLK = S_ / 32;               // 64 key blocks
    for (int it = 0; it < NBLK; ++it) {
        const int kb  = it * 32;
        const int cur = it & 1;
        if (it + 1 < NBLK) {
            issue_tiles(cur ^ 1, kb + 32);
            asm volatile("s_wait_asynccnt 4" ::: "memory");   // drain prev, keep 4 in flight
        } else {
            asm volatile("s_wait_asynccnt 0" ::: "memory");
        }
        __syncthreads();                        // all waves' tiles visible

        const unsigned short* kTile = kT[cur];
        const unsigned short* vTile = vT[cur];

        f32x8 s0 = {0,0,0,0,0,0,0,0};
        f32x8 s1 = {0,0,0,0,0,0,0,0};
        s0 = wmma_bf16(aQ0, load_b_frag(kTile, D_, 0,  0),  s0);
        s0 = wmma_bf16(aQ1, load_b_frag(kTile, D_, 0,  32), s0);
        s1 = wmma_bf16(aQ0, load_b_frag(kTile, D_, 16, 0),  s1);
        s1 = wmma_bf16(aQ1, load_b_frag(kTile, D_, 16, 32), s1);

        const float pen0 = -1.0e6f * (1.0f - mrow[kb + nloc]);
        const float pen1 = -1.0e6f * (1.0f - mrow[kb + 16 + nloc]);

        float nm[8];
#pragma unroll
        for (int r = 0; r < 8; ++r) {
            s0[r] = s0[r] * 0.125f + pen0;      // 1/sqrt(64)
            s1[r] = s1[r] * 0.125f + pen1;
            float t = fmaxf(s0[r], s1[r]);
#pragma unroll
            for (int xm = 1; xm < 16; xm <<= 1) // row max across 16 lanes
                t = fmaxf(t, __shfl_xor(t, xm, 32));
            nm[r] = fmaxf(t, rm[r]);
        }
#pragma unroll
        for (int r = 0; r < 8; ++r) {
            const float corr = __expf(rm[r] - nm[r]);
            rm[r] = nm[r];
            rs[r] *= corr;
#pragma unroll
            for (int j = 0; j < 4; ++j) ctx[j][r] *= corr;
        }
#pragma unroll
        for (int r = 0; r < 8; ++r) {
            s0[r] = __expf(s0[r] - nm[r]);
            s1[r] = __expf(s1[r] - nm[r]);
            float ls = s0[r] + s1[r];
#pragma unroll
            for (int xm = 1; xm < 16; xm <<= 1) // row sum across 16 lanes
                ls += __shfl_xor(ls, xm, 32);
            rs[r] += ls;
        }

        // P tile: C layout (lane=N, elem=M) -> LDS -> A layout (lane=M, elem=K)
#pragma unroll
        for (int r = 0; r < 8; ++r) {
            myP[(mbase + r) * 32 + nloc]      = f2bf(s0[r]);
            myP[(mbase + r) * 32 + 16 + nloc] = f2bf(s1[r]);
        }
        asm volatile("s_wait_dscnt 0" ::: "memory");

        {
            const int rr  = lane & 15;
            const int kb2 = (lane >> 4) * 8;
            const unsigned short* p = myP + rr * 32;
            uint4 lo = *reinterpret_cast<const uint4*>(p + kb2);
            uint4 hi = *reinterpret_cast<const uint4*>(p + 16 + kb2);
            const bf16x16 aP = frag_from(lo, hi);
#pragma unroll
            for (int j = 0; j < 4; ++j) {
                bf16x16 bv = load_b_frag(vTile, 32, j * 16, 0);   // rows = d
                ctx[j] = wmma_bf16(aP, bv, ctx[j]);
            }
        }
        __syncthreads();                        // done reading buffers
    }

    // epilogue: normalize and write combined-heads bf16 ctx [B*S, H*D]
#pragma unroll
    for (int r = 0; r < 8; ++r) {
        const float inv = 1.0f / rs[r];
        const int m = b * S_ + q0 + mbase + r;
#pragma unroll
        for (int j = 0; j < 4; ++j) {
            const int col = h * D_ + j * 16 + nloc;
            Ctx[(size_t)m * DIM_ + col] = f2bf(ctx[j][r] * inv);
        }
    }
}

// ---------------------------------------------------------------------------
// Host-side launch
// ---------------------------------------------------------------------------
extern "C" void kernel_launch(void* const* d_in, const int* in_sizes, int n_in,
                              void* d_out, int out_size, void* d_ws, size_t ws_size,
                              hipStream_t stream) {
    (void)in_sizes; (void)n_in; (void)out_size; (void)ws_size;

    const float* X    = (const float*)d_in[0];
    const float* mask = (const float*)d_in[1];
    const float* Wq   = (const float*)d_in[2];
    const float* bq   = (const float*)d_in[3];
    const float* Wk   = (const float*)d_in[4];
    const float* bk   = (const float*)d_in[5];
    const float* Wv   = (const float*)d_in[6];
    const float* bv   = (const float*)d_in[7];
    const float* Wo   = (const float*)d_in[8];
    const float* bo   = (const float*)d_in[9];

    uint8_t* ws = (uint8_t*)d_ws;
    const size_t MB = 1024 * 1024;
    unsigned short* Xbf = (unsigned short*)(ws + 0 * MB);   //  8 MB
    unsigned short* Wqt = (unsigned short*)(ws + 8 * MB);   //  2 MB
    unsigned short* Wkt = (unsigned short*)(ws + 10 * MB);  //  2 MB
    unsigned short* Wvt = (unsigned short*)(ws + 12 * MB);  //  2 MB
    unsigned short* Wot = (unsigned short*)(ws + 14 * MB);  //  2 MB
    unsigned short* Qh  = (unsigned short*)(ws + 16 * MB);  //  8 MB
    unsigned short* Kh  = (unsigned short*)(ws + 24 * MB);  //  8 MB
    unsigned short* Vt  = (unsigned short*)(ws + 32 * MB);  //  8 MB
    unsigned short* Ctx = (unsigned short*)(ws + 40 * MB);  //  8 MB

    // 1. converts
    cvt_f32_bf16<<<4096, 256, 0, stream>>>(X, Xbf, M_ * DIM_);
    wt_cvt_transpose<<<4096, 256, 0, stream>>>(Wq, Wqt);
    wt_cvt_transpose<<<4096, 256, 0, stream>>>(Wk, Wkt);
    wt_cvt_transpose<<<4096, 256, 0, stream>>>(Wv, Wvt);
    wt_cvt_transpose<<<4096, 256, 0, stream>>>(Wo, Wot);

    // 2. QKV projections (WMMA)
    gemm_bf16_wmma<<<1024, 128, 0, stream>>>(Xbf, Wqt, bq, Qh, 0);
    gemm_bf16_wmma<<<1024, 128, 0, stream>>>(Xbf, Wkt, bk, Kh, 0);
    gemm_bf16_wmma<<<1024, 128, 0, stream>>>(Xbf, Wvt, bv, Vt, 1);

    // 3. flash attention: one block per (head x 64-query group)
    flash_attn<<<1024, 128, 0, stream>>>(Qh, Kh, Vt, mask, Ctx);

    // 4. output projection -> fp32 d_out
    gemm_bf16_wmma<<<1024, 128, 0, stream>>>(Ctx, Wot, bo, d_out, 2);
}
